// GraphSAGE_40484361732248
// MI455X (gfx1250) — matrix operand order repaired and checked by reading, not compiled, not run on hardware
//
#include <hip/hip_runtime.h>
#include <hip/hip_bf16.h>

// ---------------------------------------------------------------------------
// GraphSAGE forward on gfx1250: bf16 WMMA (v_wmma_f32_16x16x32_bf16) GEMMs,
// gathers fused into A-fragment loads, f32 accumulation.
// Round 3: K-loop unrolling so the software-pipeline register rotation
// (a = a_next) is renamed away instead of emitted as v_mov + hazard NOPs.
// ---------------------------------------------------------------------------

typedef __attribute__((ext_vector_type(16))) __bf16 v16bf;
typedef __attribute__((ext_vector_type(8)))  __bf16 v8bf;
typedef __attribute__((ext_vector_type(8)))  float  v8f;

#define IN_CH   512
#define HID2    256      // 2*HIDDEN
#define OUT_CH  64
#define BATCH   2048
#define S0      15
#define S1      5
#define M1      (BATCH * S0)   // 30720

// ---- fragment loaders ------------------------------------------------------

// A fragment (16x32 bf16) from a gathered fp32 row: lane (m = l&15) reads
// cols [k + khalf*8 .. +7] and [k + 16 + khalf*8 .. +7], converts to bf16.
__device__ inline v16bf loadA_f32(const float* __restrict__ row, int k, int khalf) {
    const float4* p = reinterpret_cast<const float4*>(row + k + khalf * 8);
    float4 f0 = p[0], f1 = p[1];          // k .. k+7
    float4 f2 = p[4], f3 = p[5];          // k+16 .. k+23
    v16bf a;
    a[0]  = (__bf16)f0.x; a[1]  = (__bf16)f0.y; a[2]  = (__bf16)f0.z; a[3]  = (__bf16)f0.w;
    a[4]  = (__bf16)f1.x; a[5]  = (__bf16)f1.y; a[6]  = (__bf16)f1.z; a[7]  = (__bf16)f1.w;
    a[8]  = (__bf16)f2.x; a[9]  = (__bf16)f2.y; a[10] = (__bf16)f2.z; a[11] = (__bf16)f2.w;
    a[12] = (__bf16)f3.x; a[13] = (__bf16)f3.y; a[14] = (__bf16)f3.z; a[15] = (__bf16)f3.w;
    return a;
}

// Same fragment shape, but source row already bf16.
__device__ inline v16bf loadA_bf16(const __bf16* __restrict__ row, int k, int khalf) {
    const __bf16* p = row + k + khalf * 8;
    v8bf lo = *reinterpret_cast<const v8bf*>(p);        // 16B aligned
    v8bf hi = *reinterpret_cast<const v8bf*>(p + 16);
    v16bf a;
#pragma unroll
    for (int i = 0; i < 8; ++i) { a[i] = lo[i]; a[8 + i] = hi[i]; }
    return a;
}

// B fragment (32x16 bf16) from W^T [N][K]: lane (n = l&15) reads 8 bf16 at
// WT[n_tile_row][k + khalf*8] and 8 at WT[n_tile_row][k + 16 + khalf*8].
__device__ inline v16bf loadB(const __bf16* __restrict__ wt_row, int k, int khalf) {
    const __bf16* p = wt_row + k + khalf * 8;           // 16B aligned
    v8bf lo = *reinterpret_cast<const v8bf*>(p);
    v8bf hi = *reinterpret_cast<const v8bf*>(p + 16);
    v16bf b;
#pragma unroll
    for (int i = 0; i < 8; ++i) { b[i] = lo[i]; b[8 + i] = hi[i]; }
    return b;
}

// ---- weight convert + transpose: wt[n*K + k] = bf16(w[k*N + n]) ------------
__global__ __launch_bounds__(256) void wconv_kernel(const float* __restrict__ w,
                                                    __bf16* __restrict__ wt,
                                                    int K, int N) {
    int id = blockIdx.x * blockDim.x + threadIdx.x;
    if (id >= K * N) return;
    int n = id / K;
    int k = id - n * K;
    wt[id] = (__bf16)w[(size_t)k * N + n];
}

// ---- gather + mean over sampled neighbors: out[row][:] = mean_j x[nbr] -----
// grid.x = #rows, block = 128 threads, each thread owns 4 consecutive cols.
struct alignas(8) bf16x4 { __bf16 v[4]; };

__global__ __launch_bounds__(128) void gather_mean_kernel(const float* __restrict__ x,
                                                          const int* __restrict__ nbr,
                                                          int nsamp, float inv,
                                                          __bf16* __restrict__ out) {
    int row = blockIdx.x;
    int c = threadIdx.x * 4;
    float4 acc = make_float4(0.f, 0.f, 0.f, 0.f);
    const int* base = nbr + (size_t)row * nsamp;
    for (int j = 0; j < nsamp; ++j) {
        const float4* p = reinterpret_cast<const float4*>(x + (size_t)base[j] * IN_CH + c);
        float4 v = *p;
        acc.x += v.x; acc.y += v.y; acc.z += v.z; acc.w += v.w;
    }
    bf16x4 o;
    o.v[0] = (__bf16)(acc.x * inv);
    o.v[1] = (__bf16)(acc.y * inv);
    o.v[2] = (__bf16)(acc.z * inv);
    o.v[3] = (__bf16)(acc.w * inv);
    *reinterpret_cast<bf16x4*>(out + (size_t)row * IN_CH + c) = o;
}

// ---- layer-0 half-GEMM: out[:, colbase:colbase+128] =
//          relu((GATHER ? x[idx] : meanA) @ W^T + b0[colbase:]) as bf16.
// Block = 8 waves, each wave a 16-row strip (block covers 128 rows).
// K = 512, N = 128 (8 WMMA tiles). M must be a multiple of 128.
template <bool GATHER>
__global__ __launch_bounds__(256) void sage_layer0_kernel(
    const float*  __restrict__ x,       // [N_NODES, 512]     (GATHER)
    const int*    __restrict__ idx,     // [M]                (GATHER)
    const __bf16* __restrict__ meanA,   // [M, 512]           (!GATHER)
    const __bf16* __restrict__ wt,      // [128, 512]  W^T
    const float*  __restrict__ bias,    // [256]
    int colbase,
    __bf16*       __restrict__ out)     // [M, 256]
{
    const int lane  = threadIdx.x & 31;
    const int wave  = threadIdx.x >> 5;
    const int row0  = blockIdx.x * 128 + wave * 16;
    const int mrow  = lane & 15;
    const int khalf = lane >> 4;
    const int r     = row0 + mrow;

    const float*  aF = GATHER ? (x + (size_t)idx[r] * IN_CH) : nullptr;
    const __bf16* aB = GATHER ? nullptr : (meanA + (size_t)r * IN_CH);

    v8f acc[8];
#pragma unroll
    for (int t = 0; t < 8; ++t) acc[t] = (v8f){0.f,0.f,0.f,0.f,0.f,0.f,0.f,0.f};

    // software pipeline: A(k) in hand, B(k) batched, A(k+32) prefetched.
    // unroll 4 -> the a=an rotation is renamed away inside the unrolled body.
    v16bf a = GATHER ? loadA_f32(aF, 0, khalf) : loadA_bf16(aB, 0, khalf);
#pragma unroll 4
    for (int k = 0; k < IN_CH; k += 32) {
        v16bf b[8];
#pragma unroll
        for (int t = 0; t < 8; ++t)
            b[t] = loadB(wt + (size_t)(t * 16 + mrow) * IN_CH, k, khalf);
        const int kn = (k + 32 < IN_CH) ? (k + 32) : 0;   // wrap: no OOB, no branch
        v16bf an = GATHER ? loadA_f32(aF, kn, khalf) : loadA_bf16(aB, kn, khalf);
#pragma unroll
        for (int t = 0; t < 8; ++t)
            acc[t] = __builtin_amdgcn_wmma_f32_16x16x32_bf16(
                false, a, false, b[t], (short)0, acc[t], false, false);
        a = an;
    }

#pragma unroll
    for (int t = 0; t < 8; ++t) {
        int c = colbase + t * 16 + mrow;
        float bv = bias[c];
#pragma unroll
        for (int rr = 0; rr < 8; ++rr) {
            int orow = row0 + rr + 8 * khalf;
            float v = acc[t][rr] + bv;
            v = v > 0.f ? v : 0.f;
            out[(size_t)orow * HID2 + c] = (__bf16)v;
        }
    }
}

// ---- mean over the 15 hop-1 activations per batch node ---------------------
__global__ __launch_bounds__(256) void mean_h1out_kernel(const __bf16* __restrict__ in,
                                                         __bf16* __restrict__ out) {
    int b = blockIdx.x;
    int c = threadIdx.x;             // 256 cols
    float acc = 0.f;
    const __bf16* p = in + (size_t)b * S0 * HID2 + c;
    for (int j = 0; j < S0; ++j) acc += (float)p[(size_t)j * HID2];
    out[(size_t)b * HID2 + c] = (__bf16)(acc * (1.f / 15.f));
}

// ---- layer-1 GEMM: out = A1 @ Ws1 + A2 @ Wn1 + b1 (fp32 out, K=256, N=64) --
// Wave w: rows = blk*128 + w*16, 4 WMMA tiles across N, both matmuls fused.
__global__ __launch_bounds__(256) void sage_layer1_kernel(
    const __bf16* __restrict__ A1,    // [2048, 256] h0 activations
    const __bf16* __restrict__ A2,    // [2048, 256] mean of hop-1 activations
    const __bf16* __restrict__ wtS,   // [64, 256]
    const __bf16* __restrict__ wtN,   // [64, 256]
    const float*  __restrict__ bias,  // [64]
    float*        __restrict__ out)   // [2048, 64]
{
    const int lane  = threadIdx.x & 31;
    const int wave  = threadIdx.x >> 5;
    const int row0  = blockIdx.x * 128 + wave * 16;
    const int mrow  = lane & 15;
    const int khalf = lane >> 4;

    const __bf16* a1p = A1 + (size_t)(row0 + mrow) * HID2;
    const __bf16* a2p = A2 + (size_t)(row0 + mrow) * HID2;

    v8f acc[4];
#pragma unroll
    for (int t = 0; t < 4; ++t) acc[t] = (v8f){0.f,0.f,0.f,0.f,0.f,0.f,0.f,0.f};

    v16bf a1 = loadA_bf16(a1p, 0, khalf);
    v16bf a2 = loadA_bf16(a2p, 0, khalf);
#pragma unroll 2
    for (int k = 0; k < HID2; k += 32) {
        v16bf bS[4], bN[4];
#pragma unroll
        for (int t = 0; t < 4; ++t) {
            bS[t] = loadB(wtS + (size_t)(t * 16 + mrow) * HID2, k, khalf);
            bN[t] = loadB(wtN + (size_t)(t * 16 + mrow) * HID2, k, khalf);
        }
        const int kn = (k + 32 < HID2) ? (k + 32) : 0;
        v16bf a1n = loadA_bf16(a1p, kn, khalf);
        v16bf a2n = loadA_bf16(a2p, kn, khalf);
#pragma unroll
        for (int t = 0; t < 4; ++t)
            acc[t] = __builtin_amdgcn_wmma_f32_16x16x32_bf16(
                false, a1, false, bS[t], (short)0, acc[t], false, false);
#pragma unroll
        for (int t = 0; t < 4; ++t)
            acc[t] = __builtin_amdgcn_wmma_f32_16x16x32_bf16(
                false, a2, false, bN[t], (short)0, acc[t], false, false);
        a1 = a1n;
        a2 = a2n;
    }

#pragma unroll
    for (int t = 0; t < 4; ++t) {
        int c = t * 16 + mrow;
        float bv = bias[c];
#pragma unroll
        for (int rr = 0; rr < 8; ++rr) {
            int orow = row0 + rr + 8 * khalf;
            out[(size_t)orow * OUT_CH + c] = acc[t][rr] + bv;
        }
    }
}

// ---------------------------------------------------------------------------

extern "C" void kernel_launch(void* const* d_in, const int* in_sizes, int n_in,
                              void* d_out, int out_size, void* d_ws, size_t ws_size,
                              hipStream_t stream) {
    (void)in_sizes; (void)n_in; (void)out_size; (void)ws_size;

    const float* x     = (const float*)d_in[0];
    const int*   nodes = (const int*)  d_in[1];
    const int*   nb0   = (const int*)  d_in[2];
    const int*   nb1   = (const int*)  d_in[3];
    const float* w_s0  = (const float*)d_in[4];
    const float* w_n0  = (const float*)d_in[5];
    const float* b0    = (const float*)d_in[6];
    const float* w_s1  = (const float*)d_in[7];
    const float* w_n1  = (const float*)d_in[8];
    const float* b1    = (const float*)d_in[9];
    float* out = (float*)d_out;

    // workspace carve-out (256B aligned)
    char* p = (char*)d_ws;
    auto carve = [&](size_t bytes) -> char* {
        char* r = p;
        p += (bytes + 255) & ~(size_t)255;
        return r;
    };
    __bf16* wtS0   = (__bf16*)carve((size_t)128 * IN_CH * 2);   // W_self0^T
    __bf16* wtN0   = (__bf16*)carve((size_t)128 * IN_CH * 2);   // W_neigh0^T
    __bf16* wtS1   = (__bf16*)carve((size_t)OUT_CH * HID2 * 2); // W_self1^T
    __bf16* wtN1   = (__bf16*)carve((size_t)OUT_CH * HID2 * 2); // W_neigh1^T
    __bf16* m2     = (__bf16*)carve((size_t)M1 * IN_CH * 2);    // mean(h2)  [30720,512]
    __bf16* m1     = (__bf16*)carve((size_t)BATCH * IN_CH * 2); // mean(h1)  [2048,512]
    __bf16* h1out  = (__bf16*)carve((size_t)M1 * HID2 * 2);     // hop-1 act [30720,256]
    __bf16* h0out  = (__bf16*)carve((size_t)BATCH * HID2 * 2);  // hop-0 act [2048,256]
    __bf16* h1mean = (__bf16*)carve((size_t)BATCH * HID2 * 2);  // [2048,256]

    // 1) weight convert + transpose (tiny)
    wconv_kernel<<<(IN_CH * 128 + 255) / 256, 256, 0, stream>>>(w_s0, wtS0, IN_CH, 128);
    wconv_kernel<<<(IN_CH * 128 + 255) / 256, 256, 0, stream>>>(w_n0, wtN0, IN_CH, 128);
    wconv_kernel<<<(HID2 * OUT_CH + 255) / 256, 256, 0, stream>>>(w_s1, wtS1, HID2, OUT_CH);
    wconv_kernel<<<(HID2 * OUT_CH + 255) / 256, 256, 0, stream>>>(w_n1, wtN1, HID2, OUT_CH);

    // 2) neighbor means (the big HBM gathers)
    gather_mean_kernel<<<M1,    128, 0, stream>>>(x, nb1, S1, 1.f / (float)S1, m2);
    gather_mean_kernel<<<BATCH, 128, 0, stream>>>(x, nb0, S0, 1.f / (float)S0, m1);

    // 3) layer-0 GEMMs: self half (fused gather) and neigh half, per hop
    sage_layer0_kernel<true ><<<M1 / 128,    256, 0, stream>>>(x, nb0,   nullptr, wtS0, b0, 0,   h1out);
    sage_layer0_kernel<false><<<M1 / 128,    256, 0, stream>>>(nullptr, nullptr, m2, wtN0, b0, 128, h1out);
    sage_layer0_kernel<true ><<<BATCH / 128, 256, 0, stream>>>(x, nodes, nullptr, wtS0, b0, 0,   h0out);
    sage_layer0_kernel<false><<<BATCH / 128, 256, 0, stream>>>(nullptr, nullptr, m1, wtN0, b0, 128, h0out);

    // 4) mean of hop-1 activations, then layer-1 GEMM to fp32 output
    mean_h1out_kernel<<<BATCH, 256, 0, stream>>>(h1out, h1mean);
    sage_layer1_kernel<<<BATCH / 128, 256, 0, stream>>>(h0out, h1mean, wtS1, wtN1, b1, out);
}
